// DenseGeometryEdgeHead_43078521979434
// MI455X (gfx1250) — matrix operand order repaired and checked by reading, not compiled, not run on hardware
//
#include <hip/hip_runtime.h>
#include <hip/hip_bf16.h>

typedef __attribute__((ext_vector_type(16))) _Float16 v16h;
typedef __attribute__((ext_vector_type(8)))  _Float16 v8h;
typedef __attribute__((ext_vector_type(8)))  float    v8f;

#define BB   4
#define QQ   1024
#define DD   256
#define HH   64

__device__ __forceinline__ float silu_f(float x) {
    return x / (1.0f + __expf(-x));
}

// ---------------------------------------------------------------------------
// Workspace layout (halves):
//   qh     [B*Q*D]                    @ 0          (2,097,152 B)
//   qWh    [B*Q*D]                    @ 2097152    (2,097,152 B)
//   WhBpk  [8 kt][16 nt][32 ln][16 h] @ 4194304    (  131,072 B)
//   w1Bpk  [4 f][32 ln][16 h]         @ 4325376    (    4,096 B)
//   w2Bpk  [2 ks][4 f][32 ln][16 h]   @ 4329472    (    8,192 B)
// B-fragment layout: lane ln holds column n = ln&15, K rows kb..kb+15
// (kb = (ln>>4)*16), 16 contiguous halves -> two b128 loads per fragment.
// ---------------------------------------------------------------------------

__global__ void geo_prep_kernel(const float* __restrict__ q,
                                const float* __restrict__ W,
                                const float* __restrict__ w1,
                                const float* __restrict__ w2,
                                _Float16* __restrict__ qh,
                                _Float16* __restrict__ WhBpk,
                                _Float16* __restrict__ w1Bpk,
                                _Float16* __restrict__ w2Bpk) {
    int i = blockIdx.x * 256 + threadIdx.x;
    const int NQ  = BB * QQ * DD;       // 1048576
    const int NWB = 8 * 16 * 32 * 16;   // 65536
    const int NW1 = 4 * 32 * 16;        // 2048
    const int NW2 = 2 * 4 * 32 * 16;    // 4096
    if (i < NQ) { qh[i] = (_Float16)q[i]; return; }
    int t = i - NQ;
    if (t < NWB) {
        int h  = t & 15;
        int ln = (t >> 4) & 31;
        int nt = (t >> 9) & 15;
        int kt = t >> 13;
        int row = kt * 32 + ((ln >> 4) << 4) + h;
        int col = nt * 16 + (ln & 15);
        WhBpk[t] = (_Float16)W[row * DD + col];
        return;
    }
    t -= NWB;
    if (t < NW1) {
        int h  = t & 15;
        int ln = (t >> 4) & 31;
        int f  = t >> 9;
        int row = ((ln >> 4) << 4) + h;           // padded K: 6 real rows
        int col = f * 16 + (ln & 15);
        w1Bpk[t] = (row < 6) ? (_Float16)w1[row * HH + col] : (_Float16)0.0f;
        return;
    }
    t -= NW1;
    if (t < NW2) {
        int h  = t & 15;
        int ln = (t >> 4) & 31;
        int f  = (t >> 9) & 3;
        int ks = t >> 11;
        int row = ks * 32 + ((ln >> 4) << 4) + h;
        int col = f * 16 + (ln & 15);
        w2Bpk[t] = (_Float16)w2[row * HH + col];
    }
}

__device__ __forceinline__ v16h load_afrag(const _Float16* __restrict__ arow,
                                           int k0, int koff) {
    v8h lo = *(const v8h*)(arow + k0 + 2 * koff);
    v8h hi = *(const v8h*)(arow + k0 + 2 * koff + 8);
    v16h a;
    #pragma unroll
    for (int i = 0; i < 8; ++i) { a[i] = lo[i]; a[8 + i] = hi[i]; }
    return a;
}

// ---------------------------------------------------------------------------
// Kernel 1: qWh = f16(q @ W). One 16x16 output tile per wave.
// ---------------------------------------------------------------------------
__global__ void __launch_bounds__(256) geo_qw_kernel(const _Float16* __restrict__ qh,
                                                     const _Float16* __restrict__ WhBpk,
                                                     _Float16* __restrict__ qWh) {
    int wave = threadIdx.x >> 5;
    int lane = threadIdx.x & 31;
    int tile = blockIdx.x * 8 + wave;       // 0..4095
    int mt = tile >> 4;                     // row tile 0..255
    int nt = tile & 15;                     // col tile 0..15
    int nl = lane & 15;
    int g  = lane >> 4;
    int koff = g * 8;

    v8f acc = {0.f, 0.f, 0.f, 0.f, 0.f, 0.f, 0.f, 0.f};
    const _Float16* arow = qh + (mt * 16 + nl) * DD;
    #pragma unroll
    for (int kt = 0; kt < 8; ++kt) {
        v16h a = load_afrag(arow, kt * 32, koff);
        v16h b = *(const v16h*)(WhBpk + (((kt * 16 + nt) * 32) + lane) * 16);
        acc = __builtin_amdgcn_wmma_f32_16x16x32_f16(false, a, false, b,
                                                     (short)0, acc, false, false);
    }
    #pragma unroll
    for (int r = 0; r < 8; ++r)
        qWh[(mt * 16 + r + 8 * g) * DD + nt * 16 + nl] = (_Float16)acc[r];
}

// ---------------------------------------------------------------------------
// Kernel 2: main fused kernel. One 16x16 (qi x kj) tile per wave.
// ---------------------------------------------------------------------------
__global__ void __launch_bounds__(256, 1) geo_edge_kernel(
    const _Float16* __restrict__ qh,    const _Float16* __restrict__ qWh,
    const _Float16* __restrict__ w1Bpk, const _Float16* __restrict__ w2Bpk,
    const float* __restrict__ coords, const unsigned char* __restrict__ act,
    const float* __restrict__ b1, const float* __restrict__ b2,
    const float* __restrict__ w3, const float* __restrict__ b3,
    const float* __restrict__ bias, float* __restrict__ out) {

    // per-wave, double-buffered transpose tile (stride 72 avoids bank aliasing)
    __shared__ _Float16 hbuf[8][2][16][72];

    int wave = threadIdx.x >> 5;
    int lane = threadIdx.x & 31;
    int tile = blockIdx.x * 8 + wave;      // 0..16383
    int b    = tile >> 12;
    int rem  = tile & 4095;
    int tq   = (rem >> 6) << 4;
    int tk   = (rem & 63) << 4;
    int nl   = lane & 15;
    int g    = lane >> 4;
    int koff = g * 8;
    int kb   = g * 16;

    // --- resident weight B-fragments (contiguous pre-packed loads) ----------
    v16h w1B[4], w2B[2][4];
    #pragma unroll
    for (int f = 0; f < 4; ++f)
        w1B[f] = *(const v16h*)(w1Bpk + ((f * 32) + lane) * 16);
    #pragma unroll
    for (int ks = 0; ks < 2; ++ks)
        #pragma unroll
        for (int f = 0; f < 4; ++f)
            w2B[ks][f] = *(const v16h*)(w2Bpk + (((ks * 4 + f) * 32) + lane) * 16);

    float b1v[4], b2v[4], w3v[4];
    #pragma unroll
    for (int f = 0; f < 4; ++f) {
        b1v[f] = b1[f * 16 + nl];
        b2v[f] = b2[f * 16 + nl];
        w3v[f] = w3[f * 16 + nl];
    }
    float cadd = b3[0] + bias[0];

    // --- bilinear: acc = qW(rows tq..) @ q(rows tk..)^T  (K = 256) ----------
    v8f acc = {0.f, 0.f, 0.f, 0.f, 0.f, 0.f, 0.f, 0.f};
    {
        const _Float16* arow = qWh + ((b << 10) + tq + nl) * DD;
        const _Float16* brow = qh  + ((b << 10) + tk + nl) * DD;
        #pragma unroll
        for (int k0 = 0; k0 < DD; k0 += 32) {
            v16h a  = load_afrag(arow, k0, koff);
            v16h bv = *(const v16h*)(brow + k0 + kb);   // column n=nl, K=kb..kb+15
            acc = __builtin_amdgcn_wmma_f32_16x16x32_f16(false, a, false, bv,
                                                         (short)0, acc, false, false);
        }
    }

    // per-lane query coords (lane m < 16 owns row qi = tq+m)
    float qx = 0.f, qy = 0.f;
    if (lane < 16) {
        qx = coords[(((b << 10) + tq + nl) << 1) + 0];
        qy = coords[(((b << 10) + tq + nl) << 1) + 1];
    }

    // --- 16 chunks: chunk c = key column kj = tk + c ------------------------
    #pragma unroll 2
    for (int c = 0; c < 16; ++c) {
        int par = c & 1;
        float kx = coords[(((b << 10) + tk + c) << 1) + 0];
        float ky = coords[(((b << 10) + tk + c) << 1) + 1];

        // geo features directly in A-fragment layout (M = qi row, K = feature)
        v16h a1;
        #pragma unroll
        for (int h = 0; h < 16; ++h) a1[h] = (_Float16)0.0f;
        if (lane < 16) {
            float dx = qx - kx, dy = qy - ky;
            float d2 = dx * dx + dy * dy;
            float dist = sqrtf(d2 + 1e-8f);
            bool mz = (fabsf(dx) < 1e-6f) && (fabsf(dy) < 1e-6f);
            float dxs = mz ? 1e-6f : dx;
            float dys = mz ? 1e-6f : dy;
            float ang = atan2f(dys, dxs);
            a1[0] = (_Float16)dx;
            a1[1] = (_Float16)dy;
            a1[2] = (_Float16)dist;
            a1[3] = (_Float16)d2;
            a1[4] = (_Float16)__sinf(ang);
            a1[5] = (_Float16)__cosf(ang);
        }

        // layer1: [16 pairs x 32(pad)] @ [32 x 64] -> silu -> LDS (f16)
        #pragma unroll
        for (int f = 0; f < 4; ++f) {
            v8f z = {0.f, 0.f, 0.f, 0.f, 0.f, 0.f, 0.f, 0.f};
            z = __builtin_amdgcn_wmma_f32_16x16x32_f16(false, a1, false, w1B[f],
                                                       (short)0, z, false, false);
            #pragma unroll
            for (int r = 0; r < 8; ++r) {
                float x = z[r] + b1v[f];
                hbuf[wave][par][r + 8 * g][f * 16 + nl] = (_Float16)silu_f(x);
            }
        }

        // layer2: [16 x 64] @ [64 x 64]  (same-wave DS ops are in-order; the
        // compiler inserts the needed s_wait_dscnt before WMMA consumes a2)
        v8f h2[4];
        #pragma unroll
        for (int f = 0; f < 4; ++f)
            h2[f] = (v8f){0.f, 0.f, 0.f, 0.f, 0.f, 0.f, 0.f, 0.f};
        #pragma unroll
        for (int ks = 0; ks < 2; ++ks) {
            v16h a2;
            #pragma unroll
            for (int i = 0; i < 8; ++i) {
                a2[i]     = hbuf[wave][par][nl][ks * 32 + kb + i];
                a2[8 + i] = hbuf[wave][par][nl][ks * 32 + kb + 8 + i];
            }
            #pragma unroll
            for (int f = 0; f < 4; ++f)
                h2[f] = __builtin_amdgcn_wmma_f32_16x16x32_f16(false, a2, false,
                                                               w2B[ks][f], (short)0,
                                                               h2[f], false, false);
        }

        // layer3: silu then dot with w3 along N; xor-reduce within 16-lane group
        float part[8];
        #pragma unroll
        for (int r = 0; r < 8; ++r) {
            float s = 0.f;
            #pragma unroll
            for (int f = 0; f < 4; ++f)
                s += silu_f(h2[f][r] + b2v[f]) * w3v[f];
            part[r] = s;
        }
        #pragma unroll
        for (int m = 1; m <= 8; m <<= 1)
            #pragma unroll
            for (int r = 0; r < 8; ++r)
                part[r] += __shfl_xor(part[r], m, 32);

        // fold geo logit (+b3+bias) into the bilinear accumulator: the reduced
        // per-row values sit exactly in the lanes/regs owning column n == c.
        if (nl == c) {
            #pragma unroll
            for (int r = 0; r < 8; ++r)
                acc[r] += part[r] + cadd;
        }
    }

    // --- diagonal zero, activity mask, store --------------------------------
    const unsigned char* ab = act + (b << 10);
    unsigned char akj = ab[tk + nl];
    #pragma unroll
    for (int r = 0; r < 8; ++r) {
        int m  = r + 8 * g;
        int qi = tq + m;
        int kj = tk + nl;
        float v = acc[r];
        if (qi == kj) v = 0.0f;
        if (!(ab[qi] && akj)) v = -1.0e9f;
        out[(((b << 10) + qi) << 10) + kj] = v;
    }
}

// ---------------------------------------------------------------------------
extern "C" void kernel_launch(void* const* d_in, const int* in_sizes, int n_in,
                              void* d_out, int out_size, void* d_ws, size_t ws_size,
                              hipStream_t stream) {
    const float* q      = (const float*)d_in[0];
    const float* coords = (const float*)d_in[1];
    const unsigned char* act = (const unsigned char*)d_in[2];
    const float* W   = (const float*)d_in[3];
    const float* w1  = (const float*)d_in[4];
    const float* b1  = (const float*)d_in[5];
    const float* w2  = (const float*)d_in[6];
    const float* b2  = (const float*)d_in[7];
    const float* w3  = (const float*)d_in[8];
    const float* b3  = (const float*)d_in[9];
    const float* bias = (const float*)d_in[10];
    float* out = (float*)d_out;

    char* ws = (char*)d_ws;
    _Float16* qh    = (_Float16*)(ws);               // 2,097,152 B
    _Float16* qWh   = (_Float16*)(ws + 2097152);     // 2,097,152 B
    _Float16* WhBpk = (_Float16*)(ws + 4194304);     //   131,072 B
    _Float16* w1Bpk = (_Float16*)(ws + 4325376);     //     4,096 B
    _Float16* w2Bpk = (_Float16*)(ws + 4329472);     //     8,192 B

    // prep: 1048576 + 65536 + 2048 + 4096 = 1,120,256 elements
    geo_prep_kernel<<<(1120256 + 255) / 256, 256, 0, stream>>>(q, W, w1, w2,
                                                               qh, WhBpk, w1Bpk, w2Bpk);
    // qW: 4096 tiles, 8 waves/block
    geo_qw_kernel<<<512, 256, 0, stream>>>(qh, WhBpk, qWh);
    // main: 16384 tiles, 8 waves/block
    geo_edge_kernel<<<2048, 256, 0, stream>>>(qh, qWh, w1Bpk, w2Bpk, coords, act,
                                              b1, b2, w3, b3, bias, out);
}
